// RecallK_22273700397622
// MI455X (gfx1250) — matrix operand order repaired and checked by reading, not compiled, not run on hardware
//
#include <hip/hip_runtime.h>

#define NROWS 16384
#define DDIM  128
#define NSPLIT 4           // column-space splits per row-tile
#define TILES_PER_SPLIT ((NROWS / 16) / NSPLIT)   // 256 column tiles per wave

typedef __attribute__((ext_vector_type(16))) __bf16 v16bf;
typedef __attribute__((ext_vector_type(8)))  float  v8f;

union ABFrag { v16bf v; uint4 q[2]; };

__device__ __forceinline__ unsigned short f32_to_bf16_rne(float f) {
  union { float f; unsigned u; } cv; cv.f = f;
  unsigned u = cv.u;
  unsigned r = (u + 0x7FFFu + ((u >> 16) & 1u)) >> 16;
  return (unsigned short)r;
}

// ---------------------------------------------------------------------------
// Kernel 1: fp32 -> bf16 conversion + per-row squared L2 norms.
// One wave32 per row: each lane handles a float4 (128 = 32 lanes * 4).
// ---------------------------------------------------------------------------
__global__ __launch_bounds__(256)
void prep_bf16_norms(const float* __restrict__ x,
                     unsigned short* __restrict__ xb,
                     float* __restrict__ sq) {
  const int tid  = threadIdx.x;
  const int lane = tid & 31;
  const int row  = (blockIdx.x * 256 + tid) >> 5;
  const float4 f = ((const float4*)(x + (size_t)row * DDIM))[lane];
  uint2 pk;
  pk.x = (unsigned)f32_to_bf16_rne(f.x) | ((unsigned)f32_to_bf16_rne(f.y) << 16);
  pk.y = (unsigned)f32_to_bf16_rne(f.z) | ((unsigned)f32_to_bf16_rne(f.w) << 16);
  ((uint2*)(xb + (size_t)row * DDIM))[lane] = pk;
  float p = f.x*f.x + f.y*f.y + f.z*f.z + f.w*f.w;
  #pragma unroll
  for (int m = 1; m < 32; m <<= 1) p += __shfl_xor(p, m, 32);
  if (lane == 0) sq[row] = p;
}

// ---------------------------------------------------------------------------
// Kernel 2: fused X*X^T (bf16 WMMA, f32 acc) + per-row running argmin of
// score(i,j) = sq[j] - 2*dot(i,j), diagonal excluded.
// Block = 8 waves sharing one column stream (same split), each wave owning a
// 16-row tile. B tiles double-buffered through LDS (stage t+1 during t).
// ---------------------------------------------------------------------------
__global__ __launch_bounds__(256)
void nn_argmin_wmma(const unsigned short* __restrict__ xb,
                    const float* __restrict__ sq,
                    float* __restrict__ pmin,
                    int*   __restrict__ pidx) {
  __shared__ __align__(16) uint4 ldsB[2][256];   // 2 x (16 rows x 128 bf16) = 8 KB

  const int tid   = threadIdx.x;
  const int lane  = tid & 31;
  // Wave-uniform scalars: force into SGPRs so uniform branches stay scalar.
  const int wave  = __builtin_amdgcn_readfirstlane(tid >> 5);
  const int split = blockIdx.x >> 7;                     // 0..3 (scalar)
  const int rowTile =
      __builtin_amdgcn_readfirstlane(((blockIdx.x & 127) << 3) + wave);
  const int i0   = rowTile << 4;                         // scalar
  const int hi   = lane >> 4;      // half-wave select
  const int ln   = lane & 15;

  // Hoist the full A strip for this row tile: 16 rows x 128 K (bf16).
  ABFrag a[4];
  {
    const uint4* abase = (const uint4*)(xb + (size_t)(i0 + ln) * DDIM);
    #pragma unroll
    for (int ks = 0; ks < 4; ++ks) {
      const int k0 = ks * 32 + hi * 8;     // bf16 units, multiple of 8
      a[ks].q[0] = abase[k0 >> 3];
      a[ks].q[1] = abase[(k0 + 16) >> 3];
    }
  }

  float bestVal[8];
  int   bestIdx[8];
  #pragma unroll
  for (int v = 0; v < 8; ++v) { bestVal[v] = 3.4e38f; bestIdx[v] = 0; }

  const int t0 = split * TILES_PER_SPLIT;
  const int t1 = t0 + TILES_PER_SPLIT;

  // Prologue: stage first tile into buffer 0.
  ldsB[0][tid] = ((const uint4*)xb)[(size_t)(t0 << 4) * 16 + tid];

  for (int t = t0; t < t1; ++t) {
    const int j0  = t << 4;
    const int buf = t & 1;

    __syncthreads();   // staging of buf[t] visible; buf[t+1] free to overwrite

    // Stage next tile into the other buffer (overlaps with compute below).
    if (t + 1 < t1) {
      ldsB[buf ^ 1][tid] = ((const uint4*)xb)[(size_t)((t + 1) << 4) * 16 + tid];
      // Prefetch tile t+2 toward the caches.
      const int tp = (t + 2 < t1) ? (t + 2) : (t + 1);
      __builtin_prefetch(((const uint4*)xb) + ((size_t)(tp << 4) * 16 + tid), 0, 3);
    }

    const int jc = j0 + ln;                // this lane's column
    const float sqj = sq[jc];

    // B fragments: lane=column jc (local row ln), K = {hi*8..hi*8+7, 16+hi*8..16+hi*8+7}.
    const uint4* bbase =
        (const uint4*)((const unsigned short*)&ldsB[buf][0] + (size_t)ln * DDIM);
    ABFrag b[4];
    #pragma unroll
    for (int ks = 0; ks < 4; ++ks) {
      const int kb = ks * 32 + hi * 8;     // bf16 units, multiple of 8
      b[ks].q[0] = bbase[kb >> 3];
      b[ks].q[1] = bbase[(kb + 16) >> 3];
    }

    v8f acc = {};
    #pragma unroll
    for (int ks = 0; ks < 4; ++ks) {
      acc = __builtin_amdgcn_wmma_f32_16x16x32_bf16(
          /*neg_a=*/false, a[ks].v, /*neg_b=*/false, b[ks].v,
          /*c_mod=*/(short)0, acc, /*reuse_a=*/false, /*reuse_b=*/false);
    }

    // Fused epilogue: score = sq[j] - 2*dot ; track (min, argmin) per row.
    // j0 and i0 are SGPRs -> this is a scalar branch, taken for 1 tile in 256.
    if (j0 != i0) {
      #pragma unroll
      for (int v = 0; v < 8; ++v) {
        const float score = fmaf(-2.0f, acc[v], sqj);
        const bool better = score < bestVal[v];
        bestVal[v] = better ? score : bestVal[v];
        bestIdx[v] = better ? jc : bestIdx[v];
      }
    } else {
      // Diagonal tile: additionally mask out j == i (branch-free per lane).
      #pragma unroll
      for (int v = 0; v < 8; ++v) {
        const int row = i0 + hi * 8 + v;   // C layout: vgpr v -> M = v + 8*hi
        const float score = fmaf(-2.0f, acc[v], sqj);
        const int better = (int)(jc != row) & (int)(score < bestVal[v]);
        bestVal[v] = better ? score : bestVal[v];
        bestIdx[v] = better ? jc : bestIdx[v];
      }
    }
  }

  // Reduce (min, argmin) across the 16 lanes that share each row (branch-free).
  #pragma unroll
  for (int v = 0; v < 8; ++v) {
    float bv = bestVal[v];
    int   bi = bestIdx[v];
    #pragma unroll
    for (int m = 1; m < 16; m <<= 1) {
      const float ov = __shfl_xor(bv, m, 32);
      const int   oi = __shfl_xor(bi, m, 32);
      const bool better = ov < bv;
      bv = better ? ov : bv;
      bi = better ? oi : bi;
    }
    bestVal[v] = bv; bestIdx[v] = bi;
  }
  if (ln == 0) {
    #pragma unroll
    for (int v = 0; v < 8; ++v) {
      const int row = i0 + hi * 8 + v;
      pmin[(size_t)row * NSPLIT + split] = bestVal[v];
      pidx[(size_t)row * NSPLIT + split] = bestIdx[v];
    }
  }
}

// ---------------------------------------------------------------------------
// Kernel 3: merge the NSPLIT partial argmins per row, compare labels,
// reduce recall@1 into d_out[0]. Single block.
// ---------------------------------------------------------------------------
__global__ __launch_bounds__(256)
void recall_reduce(const float* __restrict__ pmin,
                   const int*   __restrict__ pidx,
                   const int*   __restrict__ labels,
                   float* __restrict__ out) {
  __shared__ int red[256];
  const int tid = threadIdx.x;
  int cnt = 0;
  for (int row = tid; row < NROWS; row += 256) {
    float bv = pmin[(size_t)row * NSPLIT];
    int   bi = pidx[(size_t)row * NSPLIT];
    #pragma unroll
    for (int s = 1; s < NSPLIT; ++s) {
      const float ov = pmin[(size_t)row * NSPLIT + s];
      const int   oi = pidx[(size_t)row * NSPLIT + s];
      const bool better = ov < bv;
      bv = better ? ov : bv;
      bi = better ? oi : bi;
    }
    cnt += (labels[bi] == labels[row]) ? 1 : 0;
  }
  red[tid] = cnt;
  __syncthreads();
  for (int s = 128; s > 0; s >>= 1) {
    if (tid < s) red[tid] += red[tid + s];
    __syncthreads();
  }
  if (tid == 0) out[0] = (float)red[0] / (float)NROWS;
}

// ---------------------------------------------------------------------------
extern "C" void kernel_launch(void* const* d_in, const int* in_sizes, int n_in,
                              void* d_out, int out_size, void* d_ws, size_t ws_size,
                              hipStream_t stream) {
  (void)in_sizes; (void)n_in; (void)out_size; (void)ws_size;
  const float* x      = (const float*)d_in[0];   // [16384,128] fp32
  const int*   labels = (const int*)d_in[1];     // [16384] int
  float* out = (float*)d_out;

  // Workspace layout (all 256B-aligned):
  char* ws = (char*)d_ws;
  unsigned short* xb = (unsigned short*)(ws + 0);                 // 4 MB bf16 bank
  float* sq   = (float*)(ws + (size_t)NROWS * DDIM * 2);          // 64 KB norms
  float* pmin = (float*)(ws + (size_t)NROWS * DDIM * 2 + 65536);  // 256 KB
  int*   pidx = (int*)  (ws + (size_t)NROWS * DDIM * 2 + 65536 + (size_t)NROWS * NSPLIT * 4);

  // 1) convert + norms: one wave per row, 8 rows per block
  prep_bf16_norms<<<NROWS / 8, 256, 0, stream>>>(x, xb, sq);

  // 2) WMMA GEMM + fused argmin: 1024 row tiles x 4 splits = 4096 waves
  nn_argmin_wmma<<<(NROWS / 16 / 8) * NSPLIT, 256, 0, stream>>>(xb, sq, pmin, pidx);

  // 3) final recall reduction
  recall_reduce<<<1, 256, 0, stream>>>(pmin, pidx, labels, out);
}
